// transBlocklayer3_61452392071713
// MI455X (gfx1250) — matrix operand order repaired and checked by reading, not compiled, not run on hardware
//
#include <hip/hip_runtime.h>
#include <hip/hip_bf16.h>

// ---------------------------------------------------------------------------
// transBlock/DropBlock fused dual-conv3x3 for MI455X (gfx1250, wave32, WMMA)
//   a = conv3x3(x, w_layer3); b = conv3x3(x, w_pre)
//   bm = 1 - maxpool7x7(mask_u < gamma); out = (bm==1 ? a : b) * size/sum(bm)
// Implicit GEMM: M = 512 rows (2 conv sets x 256 out-ch), N = spatial 50176,
// K = 9*256 = 2304 ordered k = r*256 + c so each 32-wide K chunk sits in one
// filter tap. x is pre-packed NHWC bf16 so the per-chunk fill is ONE b128
// load + ONE ds_store_b128 per thread. One block = full M x 64 positions;
// 8 wave32, 16 WMMAs/wave/chunk, double-buffered LDS + ping-ponged A regs,
// single barrier per chunk.
// ---------------------------------------------------------------------------

typedef __attribute__((ext_vector_type(16))) __bf16 v16bf;
typedef __attribute__((ext_vector_type(8)))  float  v8f;

#define NB   64
#define CCH  256
#define HH   28
#define WW   28
#define HWSZ (HH*WW)          // 784
#define NPOS (NB*HWSZ)        // 50176
#define KTOT (9*CCH)          // 2304
#define NKC  72               // K chunks of 32
#define TILE_P 64
#define XS_STRIDE 40          // shorts/row: 80B, 16B aligned, bank rot 20 words

// workspace offsets (bytes, 256-aligned)
#define WS_WB 0u
#define WS_XB 2359296u        // 2*256*2304*2
#define WS_BM 28049408u       // WS_XB + 50176*256*2
#define WS_SC 28250112u       // WS_BM + 50176*4

struct uint4x2 { uint4 lo, hi; };

static __device__ inline v16bf pack_frag(uint4 lo, uint4 hi) {
    union { uint4x2 u; v16bf v; } t;
    t.u.lo = lo; t.u.hi = hi;
    return t.v;
}

static __device__ inline unsigned short f32_to_bf16(float f) {
    unsigned u = __float_as_uint(f);
    u += 0x7FFFu + ((u >> 16) & 1u);          // round-to-nearest-even
    return (unsigned short)(u >> 16);
}

// --- prep: repack both weight tensors [O][C][3][3] -> bf16 [set][O][k=r*256+c]
__global__ __launch_bounds__(256) void prep_weights(
    const float* __restrict__ w0, const float* __restrict__ w1,
    unsigned short* __restrict__ wb)
{
    int idx = blockIdx.x * 256 + threadIdx.x;      // over 2*256*2304
    if (idx >= 2 * CCH * KTOT) return;
    int set = idx / (CCH * KTOT);
    int rem = idx - set * (CCH * KTOT);
    int o   = rem / KTOT;
    int k   = rem - o * KTOT;
    int r   = k >> 8;                               // k = r*256 + c
    int c   = k & 255;
    const float* w = set ? w1 : w0;
    wb[idx] = f32_to_bf16(w[(o * CCH + c) * 9 + r]);
}

// --- prep: x NCHW fp32 -> NHWC bf16 (xb[pos][c], channels contiguous)
__global__ __launch_bounds__(256) void prep_x(
    const float* __restrict__ x, unsigned short* __restrict__ xb)
{
    int idx = blockIdx.x * 256 + threadIdx.x;      // over NPOS*CCH (exact)
    int c   = idx & (CCH - 1);
    int pos = idx >> 8;
    int n   = pos / HWSZ;
    int hw  = pos - n * HWSZ;
    xb[idx] = f32_to_bf16(x[((size_t)(n * CCH + c)) * HWSZ + hw]);
}

// --- DropBlock block_mask: bm[p] = 1 - max_{7x7 SAME}(mask_u < gamma)
__global__ __launch_bounds__(256) void mask_kernel(
    const float* __restrict__ mu, float* __restrict__ bm)
{
    int p = blockIdx.x * 256 + threadIdx.x;
    if (p >= NPOS) return;
    int n = p / HWSZ;
    int hw = p - n * HWSZ;
    int h = hw / WW, w = hw - h * WW;
    const float gamma = 0.1f / 49.0f;
    float pooled = 0.0f;
    #pragma unroll
    for (int dy = -3; dy <= 3; ++dy) {
        int hh = h + dy;
        if (hh < 0 || hh >= HH) continue;
        #pragma unroll
        for (int dx = -3; dx <= 3; ++dx) {
            int ww = w + dx;
            if (ww < 0 || ww >= WW) continue;
            if (mu[n * HWSZ + hh * WW + ww] < gamma) pooled = 1.0f;
        }
    }
    bm[p] = 1.0f - pooled;
}

// --- single-block reduction: scale = NPOS / sum(bm)
__global__ __launch_bounds__(256) void scale_kernel(
    const float* __restrict__ bm, float* __restrict__ scale)
{
    __shared__ float red[256];
    float s = 0.0f;
    for (int i = threadIdx.x; i < NPOS; i += 256) s += bm[i];
    red[threadIdx.x] = s;
    __syncthreads();
    for (int off = 128; off > 0; off >>= 1) {
        if ((int)threadIdx.x < off) red[threadIdx.x] += red[threadIdx.x + off];
        __syncthreads();
    }
    if (threadIdx.x == 0) *scale = (float)NPOS / red[0];
}

// --- main fused dual-conv WMMA kernel ---------------------------------------
// grid = 784 position tiles; block = 256 threads = 8 wave32.
// M space: row = set*256 + outch. Wave w owns rows [w*64, w*64+64):
//   waves 0-3 -> layer3 (set 0), waves 4-7 -> premodel (set 1).
__global__ __launch_bounds__(256) void conv_wmma(
    const unsigned short* __restrict__ xb, const unsigned short* __restrict__ wb,
    const float* __restrict__ bm, const float* __restrict__ scale_p,
    float* __restrict__ out)
{
    __shared__ unsigned short Xs[2][TILE_P * XS_STRIDE];   // 2 x 5120 B

    const int tid  = threadIdx.x;
    const int lane = tid & 31;
    const int wave = tid >> 5;
    const int mset = wave >> 2;               // 0 = layer3, 1 = premodel
    const int chb  = (wave & 3) * 64;         // wave's channel base within set
    const int p0   = blockIdx.x * TILE_P;     // spatial tile base

    // fill-phase mapping: 64 positions x 4 channel groups of 8 (16B each)
    const int fpos = tid & 63;
    const int coff = (tid >> 6) * 8;
    const int p  = p0 + fpos;
    const int n  = p / HWSZ;
    const int hw = p - n * HWSZ;
    const int h  = hw / WW;
    const int w  = hw - h * WW;

    const unsigned short* wset = wb + (size_t)mset * (CCH * KTOT);
    const int lrow  = lane & 15;              // fragment row within 16
    const int lhalf = lane >> 4;              // K-half selector

    v8f acc[4][4];                            // [m strip][n subtile]
    #pragma unroll
    for (int m = 0; m < 4; ++m)
        #pragma unroll
        for (int s = 0; s < 4; ++s) acc[m][s] = (v8f){0,0,0,0,0,0,0,0};

    // --- x fill for one K chunk: ONE predicated 16B load (NHWC bf16)
    auto gather_x = [&](int kc, uint4 &xv) {
        const int r     = kc >> 3;
        const int cbase = (kc & 7) * 32;
        const int dy = r / 3 - 1;
        const int dx = r - (r / 3) * 3 - 1;
        const int hy = h + dy, wx = w + dx;
        const bool ok = (hy >= 0) & (hy < HH) & (wx >= 0) & (wx < WW);
        xv = (uint4){0u, 0u, 0u, 0u};
        if (ok) {
            const unsigned short* src =
                xb + ((size_t)((n * HH + hy) * WW + wx)) * CCH + cbase + coff;
            xv = *(const uint4*)src;
        }
    };

    auto store_x = [&](int buf, const uint4 &xv) {
        *(uint4*)&Xs[buf][fpos * XS_STRIDE + coff] = xv;
    };

    // --- A fragments (weights) for one K chunk, 16-bit A 16x32 lane layout
    auto load_a = [&](int kc, uint4 (&ar)[8]) {
        const int kb = kc * 32 + lhalf * 8;
        #pragma unroll
        for (int m = 0; m < 4; ++m) {
            const unsigned short* ap =
                wset + (size_t)(chb + m * 16 + lrow) * KTOT + kb;
            ar[2 * m]     = *(const uint4*)(ap);        // K = kb .. kb+7
            ar[2 * m + 1] = *(const uint4*)(ap + 16);   // K = kb+16 .. kb+23
        }
    };

    // --- prime the pipeline with chunk 0
    uint4 aA[8], aB[8];
    {
        uint4 xv;
        gather_x(0, xv);
        store_x(0, xv);
        load_a(0, aA);
        __syncthreads();
    }

    // --- one pipelined step: compute chunk kc, prefetch chunk kc+1
    auto step = [&](int kc, const uint4 (&acur)[8], uint4 (&anxt)[8]) {
        const int knext = (kc < NKC - 1) ? kc + 1 : kc;
        uint4 xv;
        gather_x(knext, xv);       // global loads in flight during compute
        load_a(knext, anxt);

        const int cur = kc & 1;
        v16bf af[4];
        #pragma unroll
        for (int m = 0; m < 4; ++m)
            af[m] = pack_frag(acur[2 * m], acur[2 * m + 1]);
        #pragma unroll
        for (int s = 0; s < 4; ++s) {
            const unsigned short* bp =
                &Xs[cur][(s * 16 + lrow) * XS_STRIDE + lhalf * 8];
            uint4 blo = *(const uint4*)(bp);
            uint4 bhi = *(const uint4*)(bp + 16);
            v16bf bfrag = pack_frag(blo, bhi);
            #pragma unroll
            for (int m = 0; m < 4; ++m)
                acc[m][s] = __builtin_amdgcn_wmma_f32_16x16x32_bf16(
                    false, af[m], false, bfrag, (short)0, acc[m][s],
                    false, false);
        }

        store_x(cur ^ 1, xv);      // fill the other buffer
        __syncthreads();           // single barrier per chunk
    };

    for (int kc = 0; kc < NKC; kc += 2) {   // 72 chunks, ping-pong A regs
        step(kc,     aA, aB);
        step(kc + 1, aB, aA);
    }

    // --- epilogue: blended, scaled store; complementary predicates per set
    const float scale = *scale_p;
    #pragma unroll
    for (int s = 0; s < 4; ++s) {
        const int pp = p0 + s * 16 + lrow;
        const int nn = pp / HWSZ;
        const int hh = pp - nn * HWSZ;
        const float m = bm[pp];
        const bool writeMe = (mset == 0) ? (m == 1.0f) : (m != 1.0f);
        if (writeMe) {
            #pragma unroll
            for (int ms = 0; ms < 4; ++ms) {
                const int o0 = chb + ms * 16 + lhalf * 8;
                float* op = out + ((size_t)(nn * CCH + o0)) * HWSZ + hh;
                #pragma unroll
                for (int v = 0; v < 8; ++v)
                    op[(size_t)v * HWSZ] = acc[ms][s][v] * scale;
            }
        }
    }
}

// ---------------------------------------------------------------------------
extern "C" void kernel_launch(void* const* d_in, const int* in_sizes, int n_in,
                              void* d_out, int out_size, void* d_ws, size_t ws_size,
                              hipStream_t stream) {
    const float* x  = (const float*)d_in[0];   // [64,256,28,28]
    const float* w0 = (const float*)d_in[1];   // [256,256,3,3] layer3
    const float* w1 = (const float*)d_in[2];   // [256,256,3,3] premodel
    const float* mu = (const float*)d_in[3];   // [64,28,28]

    unsigned short* wb = (unsigned short*)((char*)d_ws + WS_WB);
    unsigned short* xb = (unsigned short*)((char*)d_ws + WS_XB);
    float* bm    = (float*)((char*)d_ws + WS_BM);
    float* scale = (float*)((char*)d_ws + WS_SC);
    float* out   = (float*)d_out;

    prep_weights<<<(2 * CCH * KTOT + 255) / 256, 256, 0, stream>>>(w0, w1, wb);
    prep_x<<<(NPOS * CCH) / 256, 256, 0, stream>>>(x, xb);
    mask_kernel<<<(NPOS + 255) / 256, 256, 0, stream>>>(mu, bm);
    scale_kernel<<<1, 256, 0, stream>>>(bm, scale);

    conv_wmma<<<NPOS / TILE_P, 256, 0, stream>>>(xb, wb, bm, scale, out);
}
